// BAKT_52450140618902
// MI455X (gfx1250) — matrix operand order repaired and checked by reading, not compiled, not run on hardware
//
#include <hip/hip_runtime.h>

// ---------------------------------------------------------------------------
// Problem constants (match reference): B=32, S=512, D=512, H=8, DK=64
// ---------------------------------------------------------------------------
#define BATCH 32
#define SEQ   512
#define DIM   512
#define NH    8
#define DK    64

typedef __attribute__((ext_vector_type(16))) __bf16 v16bf;
typedef __attribute__((ext_vector_type(8)))  float  v8f;
typedef __attribute__((ext_vector_type(4)))  unsigned int u32x4;
typedef __attribute__((ext_vector_type(8)))  unsigned int u32x8;

enum { OUT_HEAD_BF16 = 0, OUT_HEAD_F32 = 1, OUT_PLAIN_F32 = 2 };

__device__ __forceinline__ unsigned short f2bf(float f) {
  union { float f; unsigned int u; } v; v.f = f;
  unsigned int u = v.u;
  u += 0x7FFFu + ((u >> 16) & 1u);         // round-to-nearest-even
  return (unsigned short)(u >> 16);
}
__device__ __forceinline__ unsigned int pack2(float a, float b) {
  return (unsigned int)f2bf(a) | ((unsigned int)f2bf(b) << 16);
}

union FragBF { v16bf v; uint4 q[2]; };

// ---------------------------------------------------------------------------
// CDNA5 async global->LDS helpers (ASYNCcnt-tracked DMA, CDNA5 ISA)
// ---------------------------------------------------------------------------
__device__ __forceinline__ unsigned lds_offset(const void* p) {
  // generic -> LDS addrspace: pointer value becomes the in-group byte offset
  return (unsigned)(unsigned long long)
      (const __attribute__((address_space(3))) void*)p;
}

__device__ __forceinline__ void async_ld_b128(unsigned lds_byte_off,
                                              const unsigned short* gsrc) {
  // GV mode: per-lane 16B copy, LDS dest offset in VGPR, 64-bit global vaddr
  asm volatile("global_load_async_to_lds_b128 %0, %1, off"
               :
               : "v"(lds_byte_off), "v"(gsrc)
               : "memory");
}

__device__ __forceinline__ void wait_async0() {
  asm volatile("s_wait_asynccnt 0x0" ::: "memory");
}

// ---------------------------------------------------------------------------
// CDNA5 Tensor Data Mover: one descriptor-driven 2D tile load (TENSORcnt).
// Loads a [rows=128][cols=64] bf16 tile (row stride 64 elts) into LDS.
// D# group0/group1 bitfields per CDNA5 ISA.
// ---------------------------------------------------------------------------
__device__ __forceinline__ void tdm_load_k_tile(const unsigned short* gsrc,
                                                unsigned lds_base) {
  const unsigned long long ga = (unsigned long long)gsrc;
  const u32x4 g0 = {
      1u,                                    // count=1, user desc
      lds_base,                              // lds_addr (bytes)
      (unsigned)ga,                          // global_addr[31:0]
      ((unsigned)(ga >> 32) & 0x01FFFFFFu)   // global_addr[56:32]
          | 0x80000000u                      // type=2 ("image") in [127:126]
  };
  const u32x8 g1 = {
      0x00010000u,          // wg_mask=0, data_size=1 (2B), no barrier/pad/iter
      (unsigned)DK << 16,   // tensor_dim0[15:0] = 64  (bits 79:48 low half)
      (unsigned)SEQ << 16,  // tensor_dim1[15:0] = 512 (bits 111:80 low half)
      (unsigned)DK << 16,   // tile_dim0 = 64          (bits 127:112)
      128u,                 // tile_dim1 = 128         (bits 143:128)
      (unsigned)DK,         // tensor_dim0_stride = 64 (bits 207:160 low 32)
      0u, 0u                // tensor_dim1_stride = 0 (unused for 2D)
  };
  asm volatile("tensor_load_to_lds %0, %1"
               :
               : "s"(g0), "s"(g1)
               : "memory");
}

__device__ __forceinline__ void wait_tensor0() {
  __builtin_amdgcn_s_wait_tensorcnt(0);
}

// ---------------------------------------------------------------------------
// Kernel 1/3: C[16384 x 512] = A[16384 x 512] @ W[512 x 512] + bias
// bf16 WMMA 16x16x32, fp32 accumulate. 128x128 tile per block, 8 waves.
// Output modes: head-layout bf16/f32 ([B,H,S,DK]) or plain row-major f32.
// ---------------------------------------------------------------------------
__global__ __launch_bounds__(256) void qkv_gemm_kernel(
    const float* __restrict__ A, const float* __restrict__ W,
    const float* __restrict__ bias, void* __restrict__ out, int mode) {
  __shared__ unsigned short ldsA[128 * 32];   // [m][k]  bf16, 8 KB
  __shared__ unsigned short ldsB[128 * 32];   // [n][k]  bf16 (W transposed), 8 KB

  const int tid  = threadIdx.x;
  const int lane = tid & 31;
  const int wave = tid >> 5;
  const int wm = wave & 3;        // 4 waves along M (32 rows each)
  const int wn = wave >> 2;       // 2 waves along N (64 cols each)
  const int m0 = blockIdx.x * 128;
  const int n0 = blockIdx.y * 128;

  v8f acc[2][4];
  #pragma unroll
  for (int tm = 0; tm < 2; ++tm)
    #pragma unroll
    for (int tn = 0; tn < 4; ++tn)
      #pragma unroll
      for (int i = 0; i < 8; ++i) acc[tm][tn][i] = 0.0f;

  for (int kk = 0; kk < DIM; kk += 32) {
    __syncthreads();
    // ---- stage A tile (128x32 fp32 -> bf16), thread: row tid/2, half (tid&1)
    {
      int r = tid >> 1;
      int c = (tid & 1) * 16;
      const float4* ga = (const float4*)(A + (size_t)(m0 + r) * DIM + kk + c);
      unsigned int* dst = (unsigned int*)&ldsA[r * 32 + c];
      #pragma unroll
      for (int i = 0; i < 4; ++i) {
        float4 f = ga[i];
        dst[2 * i]     = pack2(f.x, f.y);
        dst[2 * i + 1] = pack2(f.z, f.w);
      }
    }
    // ---- stage W tile transposed: ldsB[n][k]
    {
      int k  = tid >> 3;            // 0..31
      int nn = (tid & 7) * 16;      // 0..112
      const float4* gw = (const float4*)(W + (size_t)(kk + k) * DIM + n0 + nn);
      #pragma unroll
      for (int i = 0; i < 4; ++i) {
        float4 f = gw[i];
        ldsB[(nn + 4 * i + 0) * 32 + k] = f2bf(f.x);
        ldsB[(nn + 4 * i + 1) * 32 + k] = f2bf(f.y);
        ldsB[(nn + 4 * i + 2) * 32 + k] = f2bf(f.z);
        ldsB[(nn + 4 * i + 3) * 32 + k] = f2bf(f.w);
      }
    }
    // ---- prefetch next K-step tiles (global_prefetch_b8) to hide latency
    if (kk + 32 < DIM) {
      __builtin_prefetch(A + (size_t)(m0 + (tid >> 1)) * DIM + (kk + 32) +
                             (tid & 1) * 16, 0, 1);
      __builtin_prefetch(W + (size_t)(kk + 32 + (tid >> 3)) * DIM + n0 +
                             (tid & 7) * 16, 0, 1);
    }
    __syncthreads();

    // ---- assemble fragments per ISA 16-bit layouts
    FragBF fa[2], fb[4];
    const int abase = (lane < 16) ? 0 : 8;    // shorts: dwords 0..3 vs 4..7
    #pragma unroll
    for (int tm = 0; tm < 2; ++tm) {
      const uint4* p =
          (const uint4*)&ldsA[(wm * 32 + tm * 16 + (lane & 15)) * 32 + abase];
      fa[tm].q[0] = p[0];           // K 0..7  (lanes<16)  / 8..15
      fa[tm].q[1] = p[2];           // K 16..23            / 24..31
    }
    const int bbase = (lane < 16) ? 0 : 8;    // shorts: K 0..7,16..23 vs 8..15,24..31
    #pragma unroll
    for (int tn = 0; tn < 4; ++tn) {
      const uint4* p =
          (const uint4*)&ldsB[(wn * 64 + tn * 16 + (lane & 15)) * 32 + bbase];
      fb[tn].q[0] = p[0];
      fb[tn].q[1] = p[2];
    }
    #pragma unroll
    for (int tm = 0; tm < 2; ++tm)
      #pragma unroll
      for (int tn = 0; tn < 4; ++tn)
        acc[tm][tn] = __builtin_amdgcn_wmma_f32_16x16x32_bf16(
            false, fa[tm].v, false, fb[tn].v, (short)0, acc[tm][tn], false, false);
  }

  // ---- epilogue: C layout (lane 0-15: M=i, lane 16-31: M=i+8; N=lane&15)
  #pragma unroll
  for (int tm = 0; tm < 2; ++tm) {
    #pragma unroll
    for (int tn = 0; tn < 4; ++tn) {
      const int col = n0 + wn * 64 + tn * 16 + (lane & 15);
      const float bv = bias[col];
      #pragma unroll
      for (int i = 0; i < 8; ++i) {
        const int row = m0 + wm * 32 + tm * 16 + i + ((lane < 16) ? 0 : 8);
        const float val = acc[tm][tn][i] + bv;
        if (mode == OUT_PLAIN_F32) {
          ((float*)out)[(size_t)row * DIM + col] = val;
        } else {
          const int b = row >> 9, s = row & (SEQ - 1);
          const int h = col >> 6, d = col & (DK - 1);
          const size_t idx = (((size_t)(b * NH + h)) * SEQ + s) * DK + d;
          if (mode == OUT_HEAD_F32) ((float*)out)[idx] = val;
          else                      ((unsigned short*)out)[idx] = f2bf(val);
        }
      }
    }
  }
}

// ---------------------------------------------------------------------------
// Kernel 2: causal scores via bf16 WMMA + per-row top-5 + renormalized
// softmax over the kept entries + 5-row V gather.
// Block = (64 query rows, one b*H+h head). 8 waves: wave = (row band, col half).
// Q tile staged with GLOBAL_LOAD_ASYNC_TO_LDS_B128 (ASYNCcnt); K tiles staged
// with TENSOR_LOAD_TO_LDS (TDM, TENSORcnt), issued by wave 0 only; the next
// K tile's DMA is overlapped with the top-5 scan (single buffer).
// ---------------------------------------------------------------------------
__global__ __launch_bounds__(256) void attn_topk_kernel(
    const unsigned short* __restrict__ qh,   // [B,H,S,DK] bf16
    const unsigned short* __restrict__ kh,   // [B,H,S,DK] bf16
    const float* __restrict__ vh,            // [B,H,S,DK] f32
    const int* __restrict__ kidx_ptr,
    float* __restrict__ concat) {            // [B,S,D] f32
  __shared__ unsigned short lq[64 * 64];     // 8 KB  bf16 Q tile
  __shared__ unsigned short lk[128 * 64];    // 16 KB bf16 K tile
  __shared__ float          lsc[64 * 128];   // 32 KB fp32 score tile

  const int tid  = threadIdx.x;
  const int lane = tid & 31;
  const int wave = tid >> 5;
  const int qt = blockIdx.x;                 // 0..7 (64-query tiles)
  const int bh = blockIdx.y;                 // 0..255
  const int q0 = qt * 64;
  const size_t head = (size_t)bh * SEQ * DK;

  const unsigned lk_base = lds_offset(&lk[0]);

  // ---- issue DMA: Q tile via per-lane async, K tile 0 via TDM (wave 0)
  {
    int qr = tid >> 2, qc = (tid & 3) * 16;  // Q: 64 rows x 64 shorts
    const unsigned short* src = qh + head + (size_t)(q0 + qr) * DK + qc;
    const unsigned lq_off = lds_offset(&lq[qr * 64 + qc]);
    async_ld_b128(lq_off,      src);
    async_ld_b128(lq_off + 16, src + 8);
  }
  if (wave == 0) tdm_load_k_tile(kh + head, lk_base);
  wait_async0();
  wait_tensor0();
  __syncthreads();

  float tv[5];
  int   ti[5];
  #pragma unroll
  for (int i = 0; i < 5; ++i) { tv[i] = -3.0e38f; ti[i] = 0; }

  const int mt = wave >> 1;                  // 16-row band: 0..3
  const int nh = (wave & 1) * 64;            // column half

  const int ktmax = (q0 + 63) >> 7;          // last key tile needed (causal)
  for (int kt = 0; kt <= ktmax; ++kt) {
    // ---- scores = Q @ K^T (DK=64 -> two 16x16x32 WMMA steps)
    v8f sac[4];
    #pragma unroll
    for (int tn = 0; tn < 4; ++tn)
      #pragma unroll
      for (int i = 0; i < 8; ++i) sac[tn][i] = 0.0f;

    #pragma unroll
    for (int ks = 0; ks < 2; ++ks) {
      FragBF fa;
      {
        const uint4* p = (const uint4*)&lq[(mt * 16 + (lane & 15)) * 64 +
                                           ks * 32 + ((lane < 16) ? 0 : 8)];
        fa.q[0] = p[0];
        fa.q[1] = p[2];
      }
      #pragma unroll
      for (int tn = 0; tn < 4; ++tn) {
        FragBF fb;
        const uint4* p = (const uint4*)&lk[(nh + tn * 16 + (lane & 15)) * 64 +
                                           ks * 32 + ((lane < 16) ? 0 : 8)];
        fb.q[0] = p[0];
        fb.q[1] = p[2];
        sac[tn] = __builtin_amdgcn_wmma_f32_16x16x32_bf16(
            false, fa.v, false, fb.v, (short)0, sac[tn], false, false);
      }
    }

    // ---- scale (1/sqrt(64)) + strictly-causal mask, spill to LDS
    #pragma unroll
    for (int tn = 0; tn < 4; ++tn) {
      const int cl = nh + tn * 16 + (lane & 15);
      const int kg = kt * 128 + cl;
      #pragma unroll
      for (int i = 0; i < 8; ++i) {
        const int rl = mt * 16 + i + ((lane < 16) ? 0 : 8);
        const int qg = q0 + rl;
        float v = sac[tn][i] * 0.125f;
        if (kg >= qg) v = -1.0e30f;          // can only peek strictly past
        lsc[rl * 128 + cl] = v;
      }
    }
    __syncthreads();   // lsc complete; all waves done reading lk

    // ---- overlap: TDM next K tile into lk while we scan lsc
    if (kt < ktmax && wave == 0)
      tdm_load_k_tile(kh + head + (size_t)(kt + 1) * 128 * DK, lk_base);

    // ---- per-row top-5 (value,index) maintained in registers
    if (tid < 64) {
      const float* row = &lsc[tid * 128];
      for (int j = 0; j < 128; ++j) {
        const float v = row[j];
        if (v > tv[4]) {
          tv[4] = v;
          ti[4] = kt * 128 + j;
          #pragma unroll
          for (int p = 4; p > 0; --p) {
            if (tv[p] > tv[p - 1]) {
              float fv = tv[p]; tv[p] = tv[p - 1]; tv[p - 1] = fv;
              int   iv = ti[p]; ti[p] = ti[p - 1]; ti[p - 1] = iv;
            }
          }
        }
      }
    }

    wait_tensor0();    // wave 0 blocks until the TDM tile landed; others no-op
    __syncthreads();
  }

  // ---- renormalized softmax over kept entries + V gather
  if (tid < 64) {
    const int qg = q0 + tid;
    int kidx = *kidx_ptr;
    if (kidx > 5) kidx = 5;

    float outv[DK];
    #pragma unroll
    for (int d = 0; d < DK; ++d) outv[d] = 0.0f;

    if (qg > 0) {                            // row 0 is zero-padded
      const float mx = tv[0];
      float wgt[5];
      float wsum = 0.0f;
      #pragma unroll
      for (int i = 0; i < 5; ++i) {
        const bool valid = (i < kidx) && (tv[i] > -1.0e20f);
        wgt[i] = valid ? __expf(tv[i] - mx) : 0.0f;
        wsum += wgt[i];
      }
      const float inv = 1.0f / wsum;
      #pragma unroll
      for (int i = 0; i < 5; ++i) {
        if (wgt[i] != 0.0f) {
          const float* vr = vh + head + (size_t)ti[i] * DK;
          const float wv = wgt[i] * inv;
          #pragma unroll
          for (int d = 0; d < DK; ++d) outv[d] += wv * vr[d];
        }
      }
    }

    const int b = bh >> 3, h = bh & 7;       // bh = b*H + h
    float* dst = concat + (((size_t)b * SEQ + qg) * DIM) + h * DK;
    #pragma unroll
    for (int d = 0; d < DK; ++d) dst[d] = outv[d];
  }
}

// ---------------------------------------------------------------------------
// Host launcher
// ---------------------------------------------------------------------------
extern "C" void kernel_launch(void* const* d_in, const int* in_sizes, int n_in,
                              void* d_out, int out_size, void* d_ws, size_t ws_size,
                              hipStream_t stream) {
  (void)in_sizes; (void)n_in; (void)out_size; (void)ws_size;
  const float* q   = (const float*)d_in[0];
  const float* k   = (const float*)d_in[1];
  const float* v   = (const float*)d_in[2];
  const float* w_q = (const float*)d_in[3];
  const float* b_q = (const float*)d_in[4];
  const float* w_k = (const float*)d_in[5];
  const float* b_k = (const float*)d_in[6];
  const float* w_v = (const float*)d_in[7];
  const float* b_v = (const float*)d_in[8];
  const float* w_o = (const float*)d_in[9];
  const float* b_o = (const float*)d_in[10];
  const int*   kix = (const int*)d_in[11];

  // workspace layout (96 MiB total)
  char* ws = (char*)d_ws;
  unsigned short* qh     = (unsigned short*)(ws);                       // 16 MiB bf16
  unsigned short* kh     = (unsigned short*)(ws + (size_t)(16u << 20)); // 16 MiB bf16
  float*          vh     = (float*)(ws + (size_t)(32u << 20));          // 32 MiB f32
  float*          concat = (float*)(ws + (size_t)(64u << 20));          // 32 MiB f32

  const dim3 gemm_grid(BATCH * SEQ / 128, DIM / 128);   // (128, 4)
  const dim3 blk(256);

  qkv_gemm_kernel<<<gemm_grid, blk, 0, stream>>>(q, w_q, b_q, (void*)qh, OUT_HEAD_BF16);
  qkv_gemm_kernel<<<gemm_grid, blk, 0, stream>>>(k, w_k, b_k, (void*)kh, OUT_HEAD_BF16);
  qkv_gemm_kernel<<<gemm_grid, blk, 0, stream>>>(v, w_v, b_v, (void*)vh, OUT_HEAD_F32);

  attn_topk_kernel<<<dim3(SEQ / 64, BATCH * NH), blk, 0, stream>>>(qh, kh, vh, kix, concat);

  qkv_gemm_kernel<<<gemm_grid, blk, 0, stream>>>(concat, w_o, b_o, d_out, OUT_PLAIN_F32);
}
